// QLoRALinear_10505490006718
// MI455X (gfx1250) — matrix operand order repaired and checked by reading, not compiled, not run on hardware
//
#include <hip/hip_runtime.h>

// ---------------------------------------------------------------------------
// QLoRA fused linear for MI455X (gfx1250, wave32, WMMA).
//   out[m,n] = sum_k x[m,k]*(q[n,k]-8)*scale[n,k/64] + bias[n]
//            + 2.0 * sum_r (sum_k x[m,k]*A[k,r]) * B[r,n]
// M=8192, N=4096, K=4096, RANK=16.
//
// Fast path (needs ~96.1 MB workspace):
//   Phase 1: bandwidth-bound converts into d_ws (x->bf16, int4 dequant->bf16 W,
//            loraA -> bf16 transposed [r][k]).
//   Phase 2: bf16 WMMA GEMM, 128x128 tile / 8 waves, double-buffered LDS fed
//            by gfx1250 async-to-LDS copies (global_load_async_to_lds_b128 +
//            s_wait_asynccnt) -> near-zero inner-loop VALU, matrix-pipe bound.
// Fallback: single fused kernel with in-loop dequant (previous round).
// ---------------------------------------------------------------------------

typedef __attribute__((ext_vector_type(16))) __bf16          v16bf;
typedef __attribute__((ext_vector_type(8)))  float           v8f;
typedef __attribute__((ext_vector_type(16))) unsigned short  v16u;
typedef __attribute__((ext_vector_type(8)))  unsigned short  v8u;

#define IN_F    4096
#define OUT_F   4096
#define RANK    16
#define QBLK    64
#define BM      128
#define BN      128
#define BK      32
#define LDA     40      // padded LDS stride (halfwords): 80B = 5*16B, bank stride 20
#define SCALING 2.0f

__device__ __forceinline__ unsigned short f2bf(float f) {
    unsigned int u = __float_as_uint(f);
    u += 0x7FFFu + ((u >> 16) & 1u);          // round-to-nearest-even
    return (unsigned short)(u >> 16);
}

__device__ __forceinline__ v16bf frag16(const unsigned short* p0,
                                        const unsigned short* p1) {
    v8u lo = *(const v8u*)p0;                 // ds_load_b128
    v8u hi = *(const v8u*)p1;                 // ds_load_b128
    v16u u = __builtin_shufflevector(lo, hi, 0,1,2,3,4,5,6,7,
                                             8,9,10,11,12,13,14,15);
    union { v16u u; v16bf b; } c; c.u = u;
    return c.b;
}

// low 32 bits of a generic pointer into __shared__ = wave-relative LDS offset
__device__ __forceinline__ unsigned lds_off(const void* p) {
    return (unsigned)(size_t)p;
}

// CDNA5 async copy: 16B global -> LDS, tracked by ASYNCcnt (no VGPR staging).
__device__ __forceinline__ void async_copy_b128(unsigned lds_addr, const void* gaddr) {
    asm volatile("global_load_async_to_lds_b128 %0, %1, off"
                 :: "v"(lds_addr), "v"(gaddr) : "memory");
}
__device__ __forceinline__ void wait_async0() {
    asm volatile("s_wait_asynccnt 0x0" ::: "memory");
}

// ======================= Phase 1: conversion kernels =======================

__global__ __launch_bounds__(256)
void cvt_x_bf16_kernel(const float* __restrict__ x, unsigned short* __restrict__ xb) {
    int i = blockIdx.x * 256 + threadIdx.x;       // float4 chunk id
    float4 v = ((const float4*)x)[i];
    union { unsigned short h[4]; uint2 u; } pk;
    pk.h[0] = f2bf(v.x); pk.h[1] = f2bf(v.y);
    pk.h[2] = f2bf(v.z); pk.h[3] = f2bf(v.w);
    ((uint2*)xb)[i] = pk.u;
}

__global__ __launch_bounds__(256)
void dequant_w_bf16_kernel(const int* __restrict__ qw, const float* __restrict__ scales,
                           unsigned short* __restrict__ wb) {
    int i = blockIdx.x * 256 + threadIdx.x;       // int4 chunk id (4 codes)
    int4 q  = ((const int4*)qw)[i];
    int e0  = i << 2;                             // first element index
    int row = e0 >> 12;                           // /4096
    int blk = (e0 & 4095) >> 6;                   // quant block (chunk never straddles)
    float sc = scales[row * (IN_F / QBLK) + blk];
    union { unsigned short h[4]; uint2 u; } pk;
    pk.h[0] = f2bf((float)(q.x - 8) * sc);
    pk.h[1] = f2bf((float)(q.y - 8) * sc);
    pk.h[2] = f2bf((float)(q.z - 8) * sc);
    pk.h[3] = f2bf((float)(q.w - 8) * sc);
    ((uint2*)wb)[i] = pk.u;
}

__global__ __launch_bounds__(256)
void cvt_loraA_T_kernel(const float* __restrict__ loraA, unsigned short* __restrict__ laT) {
    int i = blockIdx.x * 256 + threadIdx.x;       // over RANK*IN_F, output-linear
    int r = i >> 12;                              // /4096
    int k = i & 4095;
    laT[i] = f2bf(loraA[(size_t)k * RANK + r]);   // laT[r][k] = A[k][r]
}

// ================= Phase 2: bf16 GEMM with async-to-LDS ====================

__global__ __launch_bounds__(256)
void qlora_gemm_async_kernel(const unsigned short* __restrict__ xb,
                             const unsigned short* __restrict__ wb,
                             const unsigned short* __restrict__ laT,
                             const float* __restrict__ bias,
                             const float* __restrict__ loraB,
                             float*       __restrict__ out)
{
    __shared__ unsigned short As[2][BM][LDA];     // x tile (bf16)
    __shared__ unsigned short Bs[2][BN][LDA];     // W tile (bf16)
    __shared__ unsigned short La[2][RANK][LDA];   // loraA tile [r][k]
    __shared__ unsigned short Xa[BM][LDA];        // x@A (scaled), k-padded to 32
    __shared__ unsigned short Lb[BN][16];         // loraB tile transposed [n][r]
    __shared__ float          BiasS[BN];

    const int tid   = threadIdx.x;
    const int lane  = tid & 31;
    const int wave  = tid >> 5;
    const int wm    = wave >> 1;                  // 0..3 : 32-row band
    const int wn    = wave & 1;                   // 0..1 : 64-col band
    const int mbase = blockIdx.y * BM;
    const int nbase = blockIdx.x * BN;
    const int l16   = lane & 15;
    const int khalf = (lane >> 4) << 3;           // unused (0 or 8); A frags also use kk
    const int kk    = (lane >> 4) << 4;           // B frag chunk: 0 or 16

    // epilogue constants (ds stores; consumed only after many barriers)
    for (int e = tid; e < BN * RANK; e += 256) {
        int n = e >> 4, r = e & 15;
        Lb[n][r] = f2bf(loraB[(size_t)r * OUT_F + nbase + n]);
    }
    if (tid < BN) BiasS[tid] = bias[nbase + tid];

    v8f zero = {};
    v8f acc[2][4];
#pragma unroll
    for (int mf = 0; mf < 2; ++mf)
#pragma unroll
        for (int nf = 0; nf < 4; ++nf) acc[mf][nf] = zero;
    v8f xacc = zero;

    // issue async copies for one K-slice into stage s (no VGPR data path)
    auto issue_async = [&](int kb, int s) {
#pragma unroll
        for (int i = 0; i < 2; ++i) {
            int c   = tid + 256 * i;              // 512 chunks of 16B per tile
            int row = c >> 2;
            int col = (c & 3) << 3;
            async_copy_b128(lds_off(&As[s][row][col]),
                            xb + (size_t)(mbase + row) * IN_F + kb + col);
            async_copy_b128(lds_off(&Bs[s][row][col]),
                            wb + (size_t)(nbase + row) * IN_F + kb + col);
        }
        if (tid < 64) {                           // loraA slice: 16 rows x 32 k
            int r   = tid >> 2;
            int col = (tid & 3) << 3;
            async_copy_b128(lds_off(&La[s][r][col]),
                            laT + (size_t)r * IN_F + kb + col);
        }
    };

    auto compute = [&](int s) {
        v16bf afrag[2];
#pragma unroll
        for (int mf = 0; mf < 2; ++mf) {
            int m = wm * 32 + mf * 16 + l16;
            afrag[mf] = frag16(&As[s][m][kk], &As[s][m][kk + 8]);
        }
#pragma unroll
        for (int nf = 0; nf < 4; ++nf) {
            int n = wn * 64 + nf * 16 + l16;
            v16bf bfrag = frag16(&Bs[s][n][kk], &Bs[s][n][kk + 8]);
#pragma unroll
            for (int mf = 0; mf < 2; ++mf)
                acc[mf][nf] = __builtin_amdgcn_wmma_f32_16x16x32_bf16(
                    false, afrag[mf], false, bfrag, (short)0, acc[mf][nf],
                    false, false);
        }
        {   // fused x@A: wave owns rows [16*wave, 16*wave+16)
            int m = wave * 16 + l16;
            v16bf axa = frag16(&As[s][m][kk], &As[s][m][kk + 8]);
            v16bf bla = frag16(&La[s][l16][kk], &La[s][l16][kk + 8]);
            xacc = __builtin_amdgcn_wmma_f32_16x16x32_bf16(
                false, axa, false, bla, (short)0, xacc, false, false);
        }
    };

    issue_async(0, 0);
    const int KT = IN_F / BK;                     // 128
    for (int kt = 0; kt < KT; ++kt) {
        int s = kt & 1;
        wait_async0();                            // own async writes to stage s done
        __syncthreads();                          // everyone's writes visible
        if (kt + 1 < KT) issue_async((kt + 1) * BK, s ^ 1);
        compute(s);
    }

    // epilogue: out += SCALING * (x@A) @ loraB via K-padded WMMA
    {
        int row = wave * 16 + ((lane >> 4) << 3);
#pragma unroll
        for (int v = 0; v < 8; ++v) {
            Xa[row + v][l16]      = f2bf(xacc[v] * SCALING);
            Xa[row + v][l16 + 16] = 0;
        }
    }
    __syncthreads();

    v16bf axa2[2];
#pragma unroll
    for (int mf = 0; mf < 2; ++mf) {
        int m = wm * 32 + mf * 16 + l16;
        axa2[mf] = frag16(&Xa[m][kk], &Xa[m][kk + 8]);
    }
#pragma unroll
    for (int nf = 0; nf < 4; ++nf) {
        int nl = wn * 64 + nf * 16 + l16;
        v8u lo = *(const v8u*)&Lb[nl][0];
        v8u hi = *(const v8u*)&Lb[nl][8];
        v16u u = __builtin_shufflevector(lo, hi, 0,1,2,3,4,5,6,7,
                                                 8,9,10,11,12,13,14,15);
        if (lane >= 16) {
#pragma unroll
            for (int i = 0; i < 16; ++i) u[i] = 0;
        }
        union { v16u u; v16bf b; } bc; bc.u = u;
#pragma unroll
        for (int mf = 0; mf < 2; ++mf)
            acc[mf][nf] = __builtin_amdgcn_wmma_f32_16x16x32_bf16(
                false, axa2[mf], false, bc.b, (short)0, acc[mf][nf],
                false, false);
    }

#pragma unroll
    for (int mf = 0; mf < 2; ++mf) {
        int mrow0 = mbase + wm * 32 + mf * 16 + ((lane >> 4) << 3);
#pragma unroll
        for (int nf = 0; nf < 4; ++nf) {
            int ncol = wn * 64 + nf * 16 + l16;
            float bv = BiasS[ncol];
            float* o = out + (size_t)mrow0 * OUT_F + nbase + ncol;
#pragma unroll
            for (int v = 0; v < 8; ++v)
                o[(size_t)v * OUT_F] = acc[mf][nf][v] + bv;
        }
    }
}

// ============== Fallback: fused kernel (in-loop dequant) ===================

__global__ __launch_bounds__(256)
void qlora_wmma_fused_kernel(const float* __restrict__ x,
                             const int*   __restrict__ qw,
                             const float* __restrict__ scales,
                             const float* __restrict__ bias,
                             const float* __restrict__ loraA,
                             const float* __restrict__ loraB,
                             float*       __restrict__ out)
{
    __shared__ unsigned short As[2][BM][LDA];
    __shared__ unsigned short Bs[2][BN][LDA];
    __shared__ unsigned short La[2][RANK][LDA];
    __shared__ unsigned short Xa[BM][LDA];
    __shared__ unsigned short Lb[BN][16];
    __shared__ float          BiasS[BN];

    const int tid   = threadIdx.x;
    const int lane  = tid & 31;
    const int wave  = tid >> 5;
    const int wm    = wave >> 1;
    const int wn    = wave & 1;
    const int mbase = blockIdx.y * BM;
    const int nbase = blockIdx.x * BN;
    const int l16   = lane & 15;
    const int khalf = (lane >> 4) << 3;
    const int kk    = (lane >> 4) << 4;

    for (int e = tid; e < BN * RANK; e += 256) {
        int n = e >> 4, r = e & 15;
        Lb[n][r] = f2bf(loraB[(size_t)r * OUT_F + nbase + n]);
    }
    if (tid < BN) BiasS[tid] = bias[nbase + tid];

    v8f zero = {};
    v8f acc[2][4];
#pragma unroll
    for (int mf = 0; mf < 2; ++mf)
#pragma unroll
        for (int nf = 0; nf < 4; ++nf) acc[mf][nf] = zero;
    v8f xacc = zero;

    auto load_tiles = [&](int kb, int s) {
#pragma unroll
        for (int i = 0; i < 4; ++i) {
            int c   = tid + 256 * i;
            int row = c >> 3;
            int col = (c & 7) << 2;
            float4 v = *(const float4*)(x + (size_t)(mbase + row) * IN_F + kb + col);
            union { unsigned short h[4]; uint2 u; } pk;
            pk.h[0] = f2bf(v.x); pk.h[1] = f2bf(v.y);
            pk.h[2] = f2bf(v.z); pk.h[3] = f2bf(v.w);
            *(uint2*)&As[s][row][col] = pk.u;
        }
#pragma unroll
        for (int i = 0; i < 4; ++i) {
            int c   = tid + 256 * i;
            int row = c >> 3;
            int col = (c & 7) << 2;
            int4 q  = *(const int4*)(qw + (size_t)(nbase + row) * IN_F + kb + col);
            float sc = scales[(nbase + row) * (IN_F / QBLK) + ((kb + col) >> 6)];
            union { unsigned short h[4]; uint2 u; } pk;
            pk.h[0] = f2bf((float)(q.x - 8) * sc);
            pk.h[1] = f2bf((float)(q.y - 8) * sc);
            pk.h[2] = f2bf((float)(q.z - 8) * sc);
            pk.h[3] = f2bf((float)(q.w - 8) * sc);
            *(uint2*)&Bs[s][row][col] = pk.u;
        }
#pragma unroll
        for (int i = 0; i < 2; ++i) {
            int e = tid + 256 * i;
            int r = e & 15, k = e >> 4;
            La[s][r][k] = f2bf(loraA[(size_t)(kb + k) * RANK + r]);
        }
    };

    auto compute = [&](int s) {
        v16bf afrag[2];
#pragma unroll
        for (int mf = 0; mf < 2; ++mf) {
            int m = wm * 32 + mf * 16 + l16;
            afrag[mf] = frag16(&As[s][m][kk], &As[s][m][kk + 8]);
        }
#pragma unroll
        for (int nf = 0; nf < 4; ++nf) {
            int n = wn * 64 + nf * 16 + l16;
            v16bf bfrag = frag16(&Bs[s][n][kk], &Bs[s][n][kk + 8]);
#pragma unroll
            for (int mf = 0; mf < 2; ++mf)
                acc[mf][nf] = __builtin_amdgcn_wmma_f32_16x16x32_bf16(
                    false, afrag[mf], false, bfrag, (short)0, acc[mf][nf],
                    false, false);
        }
        {
            int m = wave * 16 + l16;
            v16bf axa = frag16(&As[s][m][kk], &As[s][m][kk + 8]);
            v16bf bla = frag16(&La[s][l16][kk], &La[s][l16][kk + 8]);
            xacc = __builtin_amdgcn_wmma_f32_16x16x32_bf16(
                false, axa, false, bla, (short)0, xacc, false, false);
        }
    };

    load_tiles(0, 0);
    const int KT = IN_F / BK;
    for (int kt = 0; kt < KT; ++kt) {
        __syncthreads();
        int s = kt & 1;
        if (kt + 1 < KT) load_tiles((kt + 1) * BK, s ^ 1);
        compute(s);
    }

    {
        int row = wave * 16 + ((lane >> 4) << 3);
#pragma unroll
        for (int v = 0; v < 8; ++v) {
            Xa[row + v][l16]      = f2bf(xacc[v] * SCALING);
            Xa[row + v][l16 + 16] = 0;
        }
    }
    __syncthreads();

    v16bf axa2[2];
#pragma unroll
    for (int mf = 0; mf < 2; ++mf) {
        int m = wm * 32 + mf * 16 + l16;
        axa2[mf] = frag16(&Xa[m][kk], &Xa[m][kk + 8]);
    }
#pragma unroll
    for (int nf = 0; nf < 4; ++nf) {
        int nl = wn * 64 + nf * 16 + l16;
        v8u lo = *(const v8u*)&Lb[nl][0];
        v8u hi = *(const v8u*)&Lb[nl][8];
        v16u u = __builtin_shufflevector(lo, hi, 0,1,2,3,4,5,6,7,
                                                 8,9,10,11,12,13,14,15);
        if (lane >= 16) {
#pragma unroll
            for (int i = 0; i < 16; ++i) u[i] = 0;
        }
        union { v16u u; v16bf b; } bc; bc.u = u;
#pragma unroll
        for (int mf = 0; mf < 2; ++mf)
            acc[mf][nf] = __builtin_amdgcn_wmma_f32_16x16x32_bf16(
                false, axa2[mf], false, bc.b, (short)0, acc[mf][nf],
                false, false);
    }

#pragma unroll
    for (int mf = 0; mf < 2; ++mf) {
        int mrow0 = mbase + wm * 32 + mf * 16 + ((lane >> 4) << 3);
#pragma unroll
        for (int nf = 0; nf < 4; ++nf) {
            int ncol = wn * 64 + nf * 16 + l16;
            float bv = BiasS[ncol];
            float* o = out + (size_t)mrow0 * OUT_F + nbase + ncol;
#pragma unroll
            for (int v = 0; v < 8; ++v)
                o[(size_t)v * OUT_F] = acc[mf][nf][v] + bv;
        }
    }
}

// ============================== launcher ===================================

extern "C" void kernel_launch(void* const* d_in, const int* in_sizes, int n_in,
                              void* d_out, int out_size, void* d_ws, size_t ws_size,
                              hipStream_t stream) {
    const float* x      = (const float*)d_in[0];
    const int*   qw     = (const int*)  d_in[1];
    const float* scales = (const float*)d_in[2];
    const float* bias   = (const float*)d_in[3];
    const float* loraA  = (const float*)d_in[4];
    const float* loraB  = (const float*)d_in[5];
    float*       out    = (float*)d_out;

    const int M = in_sizes[0] / IN_F;                       // 8192
    dim3 grid(OUT_F / BN, M / BM);
    dim3 block(256);

    const size_t xb_elems = (size_t)M * IN_F;
    const size_t wb_elems = (size_t)OUT_F * IN_F;
    const size_t la_elems = (size_t)RANK * IN_F;
    const size_t need = (xb_elems + wb_elems + la_elems) * sizeof(unsigned short);

    if (ws_size >= need) {
        unsigned short* xb  = (unsigned short*)d_ws;
        unsigned short* wb  = xb + xb_elems;
        unsigned short* laT = wb + wb_elems;
        cvt_x_bf16_kernel   <<<(unsigned)(xb_elems / 4 / 256), block, 0, stream>>>(x, xb);
        dequant_w_bf16_kernel<<<(unsigned)(wb_elems / 4 / 256), block, 0, stream>>>(qw, scales, wb);
        cvt_loraA_T_kernel  <<<(unsigned)(la_elems / 256),     block, 0, stream>>>(loraA, laT);
        qlora_gemm_async_kernel<<<grid, block, 0, stream>>>(xb, wb, laT, bias, loraB, out);
    } else {
        qlora_wmma_fused_kernel<<<grid, block, 0, stream>>>(x, qw, scales, bias,
                                                            loraA, loraB, out);
    }
}
